// NodeSelfAtten_44521630990715
// MI455X (gfx1250) — matrix operand order repaired and checked by reading, not compiled, not run on hardware
//
#include <hip/hip_runtime.h>

// ---------------- CDNA5 WMMA types ----------------
typedef __attribute__((ext_vector_type(16))) __bf16 v16bf;
typedef __attribute__((ext_vector_type(2)))  __bf16 v2bf;
typedef __attribute__((ext_vector_type(8)))  float  v8f;

#define Bsz 4
#define Nsz 4096
#define Fsz 512
#define Csz 64
#define ROWS (Bsz*Nsz)          // 16384
#define L2E 1.44269504088896340f

// float -> bf16 via native fptrunc (lowers to v_cvt*bf16* on gfx1250)
__device__ __forceinline__ unsigned short f2bfu(float x) {
    __bf16 h = (__bf16)x;
    unsigned short s; __builtin_memcpy(&s, &h, 2);
    return s;
}
// pack two floats into two bf16 (low = a); clang fuses the two fptruncs
// into a packed convert when the subtarget supports it.
__device__ __forceinline__ unsigned bfpack2(float a, float b) {
    v2bf p;
    p[0] = (__bf16)a;
    p[1] = (__bf16)b;
    unsigned u; __builtin_memcpy(&u, &p, 4);
    return u;
}

// A fragment (16x32 bf16, MxK), src row-major [m][k], row stride ld (elements).
// ISA layout: lane m=lo; VGPR v(0..3): k = hi*8 + 2v,2v+1 ; v(4..7): k = 16 + hi*8 + ...
__device__ __forceinline__ v16bf load_a_frag(const unsigned short* base, int ld, int lo, int hi) {
    const unsigned short* r = base + lo * ld;
    uint4 c0 = *(const uint4*)(r + hi * 8);
    uint4 c1 = *(const uint4*)(r + 16 + hi * 8);
    v16bf a;
    __builtin_memcpy(&a, &c0, 16);
    __builtin_memcpy((char*)&a + 16, &c1, 16);
    return a;
}

// B fragment (32x16 bf16, KxN), src stored N-major [n][k], row stride ld (elements).
// ISA layout: lane n=lo; VGPR v(0..3): k = hi*8 + 2v,2v+1 ; v(4..7): k = 16 + hi*8 + ...
__device__ __forceinline__ v16bf load_b_frag(const unsigned short* base, int ld, int lo, int hi) {
    const unsigned short* r = base + lo * ld;
    uint4 c0 = *(const uint4*)(r + hi * 8);
    uint4 c1 = *(const uint4*)(r + 16 + hi * 8);
    v16bf b;
    __builtin_memcpy(&b, &c0, 16);
    __builtin_memcpy((char*)&b + 16, &c1, 16);
    return b;
}

__device__ __forceinline__ v8f wmma_bf16(v16bf a, v16bf b, v8f c) {
    return __builtin_amdgcn_wmma_f32_16x16x32_bf16(false, a, false, b, (short)0, c, false, false);
}

// h workspace layout: [blk = globalRow/32][col 0..511][key = globalRow%32]
// so the attention kernel can stage its transposed h tile with a linear copy.
__device__ __forceinline__ size_t h_idx(int grow, int col) {
    return ((size_t)(grow >> 5) * Fsz + col) * 32 + (grow & 31);
}

// =====================================================================
// Kernel 1: fused projection  out = relu(V @ [Wf|Wg|Wh] + [bf|bg|bh])
// M=16384, K=512, Ntot=640 (64 f | 64 g | 512 h), outputs bf16.
// Grid (128, 5), block 256 (8 waves). 128x128 output tile per WG.
// =====================================================================
__global__ __launch_bounds__(256)
void NodeSelfAtten_proj_kernel(const float* __restrict__ Vin,
                               const float* __restrict__ Wf, const float* __restrict__ bfv,
                               const float* __restrict__ Wg, const float* __restrict__ bgv,
                               const float* __restrict__ Wh, const float* __restrict__ bhv,
                               unsigned short* __restrict__ fws,
                               unsigned short* __restrict__ gws,
                               unsigned short* __restrict__ hws) {
    __shared__ unsigned short A_s[128 * 32];   // V tile, row-major [m][k], bf16
    __shared__ unsigned short Wt_s[128 * 32];  // W tile, N-major  [j][k], bf16

    const int t     = threadIdx.x;
    const int lane  = t & 31, lo = lane & 15, hi = lane >> 4;
    const int w     = t >> 5;
    const int wave_m = w >> 1;          // 0..3 -> 32-row strip
    const int wave_n = w & 1;           // 0..1 -> 64-col strip
    const int mbase = blockIdx.x * 128;
    const int jbase = blockIdx.y * 128;

    v8f acc[2][4] = {};

    for (int k0 = 0; k0 < Fsz; k0 += 32) {
        __syncthreads();
        // --- stage V tile (128x32 f32 -> bf16), packed converts ---
        {
            const int tr = t >> 3;          // 0..31
            const int tc = (t & 7) * 4;     // 0..28
            #pragma unroll
            for (int rr = 0; rr < 4; ++rr) {
                const int row = tr + 32 * rr;
                float4 vv = *(const float4*)(Vin + (size_t)(mbase + row) * Fsz + k0 + tc);
                uint2 pk;
                pk.x = bfpack2(vv.x, vv.y);
                pk.y = bfpack2(vv.z, vv.w);
                *(uint2*)(A_s + row * 32 + tc) = pk;
            }
        }
        // --- stage W tile transposed ([j][k]); convert 2 k-rows per store ---
        #pragma unroll
        for (int i = 0; i < 8; ++i) {
            const int lin = t + 256 * i;           // 0..2047
            const int kp  = lin >> 7;              // 0..15 (k pair)
            const int jj  = lin & 127;
            const int j   = jbase + jj;
            const int k   = k0 + 2 * kp;
            float w0, w1;
            if (j < 64)       { w0 = Wf[(size_t)k * Csz + j];
                                w1 = Wf[(size_t)(k + 1) * Csz + j]; }
            else if (j < 128) { w0 = Wg[(size_t)k * Csz + (j - 64)];
                                w1 = Wg[(size_t)(k + 1) * Csz + (j - 64)]; }
            else              { w0 = Wh[(size_t)k * Fsz + (j - 128)];
                                w1 = Wh[(size_t)(k + 1) * Fsz + (j - 128)]; }
            *(unsigned*)(Wt_s + jj * 32 + 2 * kp) = bfpack2(w0, w1);
        }
        __syncthreads();

        // --- 8 WMMAs per K-step per wave ---
        v16bf a[2];
        #pragma unroll
        for (int mi = 0; mi < 2; ++mi)
            a[mi] = load_a_frag(A_s + (wave_m * 32 + mi * 16) * 32, 32, lo, hi);
        #pragma unroll
        for (int ni = 0; ni < 4; ++ni) {
            v16bf bb = load_b_frag(Wt_s + (wave_n * 64 + ni * 16) * 32, 32, lo, hi);
            #pragma unroll
            for (int mi = 0; mi < 2; ++mi)
                acc[mi][ni] = wmma_bf16(a[mi], bb, acc[mi][ni]);
        }
    }

    // --- epilogue: +bias, relu, cvt bf16, routed store ---
    #pragma unroll
    for (int ni = 0; ni < 4; ++ni) {
        const int j = jbase + wave_n * 64 + ni * 16 + lo;
        float bias;
        if (j < 64)       bias = bfv[j];
        else if (j < 128) bias = bgv[j - 64];
        else              bias = bhv[j - 128];
        #pragma unroll
        for (int mi = 0; mi < 2; ++mi) {
            #pragma unroll
            for (int r = 0; r < 8; ++r) {
                const int row = mbase + wave_m * 32 + mi * 16 + r + 8 * hi;
                float v = acc[mi][ni][r] + bias;
                v = fmaxf(v, 0.0f);
                const unsigned short bv = f2bfu(v);
                if (j < 64)       fws[(size_t)row * Csz + j] = bv;
                else if (j < 128) gws[(size_t)row * Csz + (j - 64)] = bv;
                else              hws[h_idx(row, j - 128)] = bv;
            }
        }
    }
}

// =====================================================================
// Kernel 2: flash attention.
//   S = f q-tile(64) @ g^T per 32-key block, online softmax, O += P @ h.
//   out = gamma * (O / l) + V
// Grid (64, 4), block 256.  Wave w: rows 16*(w>>1), cols 256*(w&1).
// =====================================================================
__global__ __launch_bounds__(256)
void NodeSelfAtten_attn_kernel(const unsigned short* __restrict__ fws,
                               const unsigned short* __restrict__ gws,
                               const unsigned short* __restrict__ hws,
                               const float* __restrict__ Vin,
                               const float* __restrict__ gamma,
                               float* __restrict__ out) {
    __shared__ unsigned short f_s[64 * 64];      // q tile [row][c]               8 KB
    __shared__ unsigned short g_s[32 * 64];      // key tile [key][c]             4 KB
    __shared__ unsigned short hT_s[512 * 32];    // h tile transposed [col][key] 32 KB
    __shared__ float          S_s[64 * 32];      // scores                        8 KB
    __shared__ unsigned short P_s[64 * 32];      // probs bf16                    4 KB
    __shared__ float          fac_s[64];
    __shared__ float          l_s[64];

    const int t      = threadIdx.x;
    const int lane   = t & 31, lo = lane & 15, hi = lane >> 4;
    const int w      = t >> 5;
    const int wave_m = w >> 1;      // 0..3 -> 16 query rows
    const int wave_f = w & 1;       // 0..1 -> 256 output cols
    const int qbase  = blockIdx.x * 64;
    const int b      = blockIdx.y;

    // stage persistent q tile (64x64 bf16 = 512 uint4, linear copy)
    {
        const uint4* src = (const uint4*)(fws + (size_t)(b * Nsz + qbase) * Csz);
        uint4* dst = (uint4*)f_s;
        dst[t]       = src[t];
        dst[t + 256] = src[t + 256];
    }

    float m_i = -1e30f, l_i = 0.0f;    // row stats (valid in lanes t<64)
    v8f acc[16] = {};                  // 16 rows x 256 cols fp32 per wave

    for (int jb = 0; jb < Nsz / 32; ++jb) {
        const int kbase = jb * 32;
        __syncthreads();   // previous P@h done before overwriting g/hT

        // --- stage g key tile (32x64 = 256 uint4, linear) ---
        {
            const uint4* src = (const uint4*)(gws + (size_t)(b * Nsz + kbase) * Csz);
            ((uint4*)g_s)[t] = src[t];
        }
        // --- stage transposed h tile: pure linear copy (layout done by proj) ---
        {
            const uint4* src = (const uint4*)(hws + (size_t)(b * 128 + jb) * Fsz * 32);
            uint4* dst = (uint4*)hT_s;
            #pragma unroll
            for (int i = 0; i < 8; ++i)
                dst[t + 256 * i] = src[t + 256 * i];
        }
        __syncthreads();

        // --- S tile: rows 16*wave_m, cols 16*wave_f, K = 64 (2 WMMAs) ---
        {
            v16bf a0 = load_a_frag(f_s + (16 * wave_m) * 64,      64, lo, hi);
            v16bf a1 = load_a_frag(f_s + (16 * wave_m) * 64 + 32, 64, lo, hi);
            v16bf b0 = load_b_frag(g_s + (16 * wave_f) * 64,      64, lo, hi);
            v16bf b1 = load_b_frag(g_s + (16 * wave_f) * 64 + 32, 64, lo, hi);
            v8f s = {};
            s = wmma_bf16(a0, b0, s);
            s = wmma_bf16(a1, b1, s);
            #pragma unroll
            for (int r = 0; r < 8; ++r)
                S_s[(16 * wave_m + r + 8 * hi) * 32 + 16 * wave_f + lo] = s[r];
        }
        __syncthreads();

        // --- online softmax (one lane per query row) ---
        if (t < 64) {
            const int row = t;
            float mx = m_i;
            #pragma unroll 8
            for (int c = 0; c < 32; ++c) mx = fmaxf(mx, S_s[row * 32 + c]);
            const float fac = exp2f((m_i - mx) * L2E);
            float sum = 0.0f;
            #pragma unroll 4
            for (int c = 0; c < 32; c += 2) {
                float p0 = exp2f((S_s[row * 32 + c]     - mx) * L2E);
                float p1 = exp2f((S_s[row * 32 + c + 1] - mx) * L2E);
                sum += p0 + p1;
                *(unsigned*)(P_s + row * 32 + c) = bfpack2(p0, p1);
            }
            l_i = l_i * fac + sum;
            m_i = mx;
            fac_s[row] = fac;
        }
        __syncthreads();

        // --- rescale accumulators by this block's factor ---
        {
            float fr[8];
            #pragma unroll
            for (int r = 0; r < 8; ++r) fr[r] = fac_s[16 * wave_m + r + 8 * hi];
            #pragma unroll
            for (int nt = 0; nt < 16; ++nt)
                #pragma unroll
                for (int r = 0; r < 8; ++r) acc[nt][r] *= fr[r];
        }

        // --- O += P @ h  (16 WMMAs per wave, K=32) ---
        {
            v16bf a = load_a_frag(P_s + (16 * wave_m) * 32, 32, lo, hi);
            #pragma unroll
            for (int nt = 0; nt < 16; ++nt) {
                v16bf bb = load_b_frag(hT_s + (256 * wave_f + nt * 16) * 32, 32, lo, hi);
                acc[nt] = wmma_bf16(a, bb, acc[nt]);
            }
        }
    }

    // --- epilogue: out = gamma * (O / l) + V ---
    if (t < 64) l_s[t] = l_i;
    __syncthreads();

    float linv[8];
    #pragma unroll
    for (int r = 0; r < 8; ++r) linv[r] = 1.0f / l_s[16 * wave_m + r + 8 * hi];

    #pragma unroll
    for (int nt = 0; nt < 16; ++nt) {
        const int col = 256 * wave_f + nt * 16 + lo;
        const float gam = gamma[col];
        #pragma unroll
        for (int r = 0; r < 8; ++r) {
            const int row = qbase + 16 * wave_m + r + 8 * hi;
            const size_t idx = ((size_t)(b * Nsz + row)) * Fsz + col;
            out[idx] = gam * (acc[nt][r] * linv[r]) + Vin[idx];
        }
    }
}

// =====================================================================
extern "C" void kernel_launch(void* const* d_in, const int* in_sizes, int n_in,
                              void* d_out, int out_size, void* d_ws, size_t ws_size,
                              hipStream_t stream) {
    (void)in_sizes; (void)n_in; (void)out_size; (void)ws_size;
    const float* V     = (const float*)d_in[0];
    const float* bf    = (const float*)d_in[2];
    const float* Wf    = (const float*)d_in[1];
    const float* Wg    = (const float*)d_in[3];
    const float* bg    = (const float*)d_in[4];
    const float* Wh    = (const float*)d_in[5];
    const float* bh    = (const float*)d_in[6];
    const float* gamma = (const float*)d_in[7];

    // workspace: f (2 MB) | g (2 MB) | h (16 MB, transposed-block layout), all bf16
    unsigned short* fws = (unsigned short*)d_ws;
    unsigned short* gws = fws + (size_t)ROWS * Csz;
    unsigned short* hws = gws + (size_t)ROWS * Csz;

    dim3 gridP(ROWS / 128, 5);          // 640 cols / 128
    NodeSelfAtten_proj_kernel<<<gridP, 256, 0, stream>>>(V, Wf, bf, Wg, bg, Wh, bh,
                                                         fws, gws, hws);

    dim3 gridA(Nsz / 64, Bsz);
    NodeSelfAtten_attn_kernel<<<gridA, 256, 0, stream>>>(fws, gws, hws, V, gamma,
                                                         (float*)d_out);
}
